// QCNetAgentEncoder_19799799234865
// MI455X (gfx1250) — compile-verified, matrix operand order checked by reading, not run on hardware
//
#include <hip/hip_runtime.h>

// ---------------------------------------------------------------------------
// QCNet agent encoder for MI455X (gfx1250).
// All dense GEMMs -> v_wmma_f32_16x16x32_bf16 (bf16 operands, f32 accum).
// B weight panel (64 cols x K) staged per-block into LDS via the Tensor Data
// Mover (TENSOR_LOAD_TO_LDS + s_wait_tensorcnt) when available; fragments
// then come from ds_load_b128. A operands row-major bf16 with rows padded to
// x64 (guard-free b128 loads). Per-edge intermediates bf16; segment softmax
// via atomics. Wave32 everywhere.
// ---------------------------------------------------------------------------

#define TT 50
#define AA 128
#define PP 256
#define NF 64

typedef __bf16 bf16;
typedef __attribute__((ext_vector_type(4)))  __bf16 bf16x4;
typedef __attribute__((ext_vector_type(8)))  __bf16 bf16x8;
typedef __attribute__((ext_vector_type(16))) __bf16 bf16x16;
typedef __attribute__((ext_vector_type(8)))  float  f32x8;
typedef unsigned int u32x4 __attribute__((ext_vector_type(4)));
typedef int          i32x4 __attribute__((ext_vector_type(4)));
typedef int          i32x8 __attribute__((ext_vector_type(8)));

#if defined(__has_builtin)
#  if __has_builtin(__builtin_amdgcn_tensor_load_to_lds) && \
      __has_builtin(__builtin_amdgcn_s_wait_tensorcnt)
#    define HAVE_TDM 1
#  else
#    define HAVE_TDM 0
#  endif
#else
#  define HAVE_TDM 0
#endif

typedef __attribute__((address_space(3))) bf16 lds_bf16_t;

// --- misc device helpers ----------------------------------------------------
static __device__ __forceinline__ float wrap_pi(float a) {
  const float PI = 3.14159265358979323846f;
  float m = fmodf(a + PI, 2.0f * PI);
  if (m < 0.0f) m += 2.0f * PI;
  return m - PI;
}
static __device__ __forceinline__ float ang_between(float cx, float cy, float nx, float ny) {
  return atan2f(cx * ny - cy * nx, cx * nx + cy * ny);
}
static __device__ __forceinline__ unsigned fenc(float f) {
  unsigned u = __float_as_uint(f);
  return (u & 0x80000000u) ? ~u : (u | 0x80000000u);
}
static __device__ __forceinline__ float fdec(unsigned u) {
  unsigned v = (u & 0x80000000u) ? (u & 0x7fffffffu) : ~u;
  return __uint_as_float(v);
}

// ---------------------------------------------------------------------------
// WMMA GEMM: C[M,N] = act( A[M,K]_bf16 * B^T[N,K]_bf16 + bias (+ C) )
// One wave computes a 16x64 C tile (4 accumulators). 4 waves / block.
// useLds: stage the block's 64xK B panel in LDS (TDM when available).
// Fragment K-map (ISA 7.12.2, 16-bit 16x32): lane group g holds K runs
// {g*8..g*8+7} and {16+g*8..16+g*8+7}  -> two contiguous b128 loads.
// ---------------------------------------------------------------------------
#define LDS_B_ELEMS (64 * 256)   // 32 KB, fits K<=256

__global__ void gemm_kernel(const bf16* __restrict__ A, const bf16* __restrict__ Bt,
                            const float* __restrict__ bias,
                            float* __restrict__ Cf, bf16* __restrict__ Cb,
                            int M, int N, int K, int ldb, int beta, int act,
                            int useLds) {
  __shared__ bf16 Bs[LDS_B_ELEMS];
  int lane = threadIdx.x & 31;
  int wv   = threadIdx.x >> 5;
  int mt   = blockIdx.x * 4 + wv;
  int row0 = mt * 16;
  int colBase = blockIdx.y * 64;
  int half = lane >> 4;
  int l15  = lane & 15;

  if (useLds) {
#if HAVE_TDM
    if (wv == 0) {   // wave-uniform: one TDM descriptor per block
      size_t gaddr = (size_t)(const void*)(Bt + (size_t)colBase * ldb);
      unsigned ldsOff = (unsigned)(size_t)(lds_bf16_t*)Bs;
      unsigned uK = (unsigned)K, uLdb = (unsigned)ldb;
      // D# group0: count=1 | lds_addr | global_addr | type=2
      u32x4 g0;
      g0[0] = 1u;
      g0[1] = ldsOff;
      g0[2] = (unsigned)(gaddr & 0xffffffffu);
      g0[3] = (unsigned)((gaddr >> 32) & 0x01ffffffu) | (2u << 30);
      // D# group1: data_size=2B; tensor_dim0=K; tensor_dim1=64;
      // tile_dim0=K; tile_dim1=64; tensor_dim0_stride=ldb (2D tile)
      i32x8 g1;
      g1[0] = (int)(1u << 16);                                   // data_size=1 (2 bytes)
      g1[1] = (int)((uK & 0xffffu) << 16);                       // tensor_dim0[15:0]
      g1[2] = (int)(((uK >> 16) & 0xffffu) | (64u << 16));       // tensor_dim0[31:16] | tensor_dim1[15:0]
      g1[3] = (int)((uK & 0xffffu) << 16);                       // tensor_dim1[31:16]=0 | tile_dim0
      g1[4] = (int)64u;                                          // tile_dim1=64 | tile_dim2=0
      g1[5] = (int)uLdb;                                         // tensor_dim0_stride[31:0]
      g1[6] = 0;                                                 // stride[47:32] | dim1_stride lo
      g1[7] = 0;
      i32x4 g2; g2[0] = 0; g2[1] = 0; g2[2] = 0; g2[3] = 0;
      i32x4 g3 = g2;
#if defined(__clang_major__) && (__clang_major__ >= 23)
      i32x8 gz; for (int z = 0; z < 8; ++z) gz[z] = 0;
      __builtin_amdgcn_tensor_load_to_lds(g0, g1, g2, g3, gz, 0);
#else
      __builtin_amdgcn_tensor_load_to_lds(g0, g1, g2, g3, 0);
#endif
      __builtin_amdgcn_s_wait_tensorcnt(0);
    }
#else
    // cooperative fallback: 128 threads copy 64 rows x K bf16 (16B chunks)
    int chunks = (64 * K) / 8;
    int kc = K / 8;
    for (int c = threadIdx.x; c < chunks; c += 128) {
      int rrow = c / kc;
      int kk = (c % kc) * 8;
      *(bf16x8*)(Bs + rrow * K + kk) =
          *(const bf16x8*)(Bt + (size_t)(colBase + rrow) * ldb + kk);
    }
#endif
    __syncthreads();
  }

  if (row0 >= M) return;              // wave-uniform exit (after the barrier)
  const bf16* Arow = A + (size_t)(row0 + l15) * K;

  f32x8 acc[4];
#pragma unroll
  for (int t = 0; t < 4; ++t) {
#pragma unroll
    for (int j = 0; j < 8; ++j) {
      float c0 = 0.0f;
      if (beta) c0 = Cf[(size_t)(row0 + j + half * 8) * N + colBase + t * 16 + l15];
      acc[t][j] = c0;
    }
  }

  for (int k0 = 0; k0 < K; k0 += 32) {
    bf16x8 a_lo = *(const bf16x8*)(Arow + k0 + half * 8);
    bf16x8 a_hi = *(const bf16x8*)(Arow + k0 + 16 + half * 8);
    bf16x16 av = __builtin_shufflevector(a_lo, a_hi,
        0, 1, 2, 3, 4, 5, 6, 7, 8, 9, 10, 11, 12, 13, 14, 15);
#pragma unroll
    for (int t = 0; t < 4; ++t) {
      bf16x8 b_lo, b_hi;
      if (useLds) {
        const bf16* BcolL = Bs + (size_t)(t * 16 + l15) * K;
        b_lo = *(const bf16x8*)(BcolL + k0 + half * 8);          // ds_load_b128
        b_hi = *(const bf16x8*)(BcolL + k0 + 16 + half * 8);
      } else {
        const bf16* Bcol = Bt + (size_t)(colBase + t * 16 + l15) * ldb;
        b_lo = *(const bf16x8*)(Bcol + k0 + half * 8);
        b_hi = *(const bf16x8*)(Bcol + k0 + 16 + half * 8);
      }
      bf16x16 bv = __builtin_shufflevector(b_lo, b_hi,
          0, 1, 2, 3, 4, 5, 6, 7, 8, 9, 10, 11, 12, 13, 14, 15);
      acc[t] = __builtin_amdgcn_wmma_f32_16x16x32_bf16(false, av, false, bv,
                                                       (short)0, acc[t], false, false);
    }
  }

#pragma unroll
  for (int t = 0; t < 4; ++t) {
#pragma unroll
    for (int j = 0; j < 8; ++j) {
      size_t rr = (size_t)(row0 + j + half * 8);
      int cc = colBase + t * 16 + l15;
      float v = acc[t][j];
      if (bias) v += bias[cc];
      if (act == 1)      v = v > 0.0f ? v : 0.0f;
      else if (act == 2) v = 1.0f / (1.0f + expf(-v));
      if (Cf) Cf[rr * N + cc] = v;
      if (Cb) Cb[rr * N + cc] = (bf16)v;
    }
  }
}

// ---------------------------------------------------------------------------
// LayerNorm over 128-wide rows; one wave per row; per-lane contiguous 4-elem
// chunks; optional relu / residual; f32 and/or bf16 outputs.
// ---------------------------------------------------------------------------
template <typename Tin>
__global__ void ln128_kernel(const Tin* __restrict__ in, const float* __restrict__ g,
                             const float* __restrict__ b, const float* __restrict__ res,
                             float* __restrict__ outf, bf16* __restrict__ outb,
                             int rows, int act) {
  int lane = threadIdx.x & 31;
  int row  = blockIdx.x * (blockDim.x >> 5) + (threadIdx.x >> 5);
  if (row >= rows) return;
  const Tin* rp = in + (size_t)row * 128 + lane * 4;
  float v[4];
  float s = 0.0f;
#pragma unroll
  for (int i = 0; i < 4; ++i) { v[i] = (float)rp[i]; s += v[i]; }
#pragma unroll
  for (int m = 1; m < 32; m <<= 1) s += __shfl_xor(s, m, 32);
  float mean = s * (1.0f / 128.0f);
  float sq = 0.0f;
#pragma unroll
  for (int i = 0; i < 4; ++i) { float d = v[i] - mean; sq += d * d; }
#pragma unroll
  for (int m = 1; m < 32; m <<= 1) sq += __shfl_xor(sq, m, 32);
  float rstd = rsqrtf(sq * (1.0f / 128.0f) + 1e-5f);
#pragma unroll
  for (int i = 0; i < 4; ++i) {
    int c = lane * 4 + i;
    float o = (v[i] - mean) * rstd * g[c] + b[c];
    if (act == 1) o = o > 0.0f ? o : 0.0f;
    if (res) o += res[(size_t)row * 128 + c];
    if (outf) outf[(size_t)row * 128 + c] = o;
    if (outb) outb[(size_t)row * 128 + c] = (bf16)o;
  }
}

// ---------------------------------------------------------------------------
// Fourier feature expansion: one wave per row, writes [cos|sin|x|0pad] (160)
// as bf16 (GEMM A operand, K padded 129 -> 160).
// ftype: 0=x_a, 1=r_t, 2=r_pl2a, 3=r_a2a
// ---------------------------------------------------------------------------
__global__ void fourier_feat_kernel(int ftype, int idim, int chunk_base, int nrows,
                                    const float* __restrict__ freqs,
                                    const float* __restrict__ pos_a,
                                    const float* __restrict__ head_a,
                                    const float* __restrict__ mv,
                                    const float* __restrict__ vel,
                                    const float* __restrict__ pos_pl,
                                    const float* __restrict__ orient_pl,
                                    const int* __restrict__ es,
                                    const int* __restrict__ ed,
                                    bf16* __restrict__ out) {
  int lane = threadIdx.x & 31;
  int r    = blockIdx.x * (blockDim.x >> 5) + (threadIdx.x >> 5);
  if (r >= nrows) return;
  int n = chunk_base + r;
  float val = 0.0f;
  if (ftype == 0) {
    float hd = head_a[n];
    float hx = cosf(hd), hy = sinf(hd);
    if (idim == 0)      { float x = mv[2*n],  y = mv[2*n+1];  val = sqrtf(x*x + y*y); }
    else if (idim == 1) { float x = mv[2*n],  y = mv[2*n+1];  val = ang_between(hx, hy, x, y); }
    else if (idim == 2) { float x = vel[2*n], y = vel[2*n+1]; val = sqrtf(x*x + y*y); }
    else                { float x = vel[2*n], y = vel[2*n+1]; val = ang_between(hx, hy, x, y); }
  } else if (ftype == 1) {
    int s = es[n], d = ed[n];
    float rx = pos_a[2*s] - pos_a[2*d], ry = pos_a[2*s+1] - pos_a[2*d+1];
    if (idim == 0)      val = sqrtf(rx*rx + ry*ry);
    else if (idim == 1) { float hd = head_a[d]; val = ang_between(cosf(hd), sinf(hd), rx, ry); }
    else if (idim == 2) val = wrap_pi(head_a[s] - head_a[d]);
    else                val = (float)(s - d);
  } else if (ftype == 2) {
    int sp = es[n], da = ed[n];
    int p = sp % PP;
    int a = da % AA, t = da / AA;
    int nd = a * TT + t;                       // (T,A) -> (A,T) flat index
    float rx = pos_pl[2*p] - pos_a[2*nd], ry = pos_pl[2*p+1] - pos_a[2*nd+1];
    float hd = head_a[nd];
    if (idim == 0)      val = sqrtf(rx*rx + ry*ry);
    else if (idim == 1) val = ang_between(cosf(hd), sinf(hd), rx, ry);
    else                val = wrap_pi(orient_pl[p] - hd);
  } else {
    int s = es[n], d = ed[n];
    int as_ = s % AA, ts = s / AA; int ns = as_ * TT + ts;
    int ad  = d % AA, td = d / AA; int nd = ad  * TT + td;
    float rx = pos_a[2*ns] - pos_a[2*nd], ry = pos_a[2*ns+1] - pos_a[2*nd+1];
    float hd = head_a[nd];
    if (idim == 0)      val = sqrtf(rx*rx + ry*ry);
    else if (idim == 1) val = ang_between(cosf(hd), sinf(hd), rx, ry);
    else                val = wrap_pi(head_a[ns] - hd);
  }
  const float twopi = 6.28318530717958647692f;
  bf16* op = out + (size_t)r * 160;
#pragma unroll
  for (int m2 = 0; m2 < 5; ++m2) {
    int j = lane + 32 * m2;
    float o;
    if (j < 64)        o = cosf(twopi * val * freqs[j]);
    else if (j < 128)  o = sinf(twopi * val * freqs[j - 64]);
    else if (j == 128) o = val;
    else               o = 0.0f;
    op[j] = (bf16)o;
  }
}

__global__ void addcat_kernel(float* __restrict__ hsum, const float* __restrict__ type_emb,
                              const int* __restrict__ atype, int chunk_base, int nrows) {
  int idx = blockIdx.x * blockDim.x + threadIdx.x;
  if (idx >= nrows * 128) return;
  int r = idx >> 7, c = idx & 127;
  int n = chunk_base + r;
  hsum[idx] += type_emb[atype[n / TT] * 128 + c];
}

// ---------------------------------------------------------------------------
// Attention edge kernels (vectorized per-lane 4-element chunks)
// ---------------------------------------------------------------------------
__global__ void edge_msg_kernel(const float* __restrict__ q, const float* __restrict__ k,
                                const float* __restrict__ v,
                                const bf16* __restrict__ kr, const bf16* __restrict__ vr,
                                const int* __restrict__ es, const int* __restrict__ ed,
                                float* __restrict__ sim, bf16* __restrict__ vj,
                                int e_base, int ne) {
  int lane = threadIdx.x & 31;
  int er   = blockIdx.x * (blockDim.x >> 5) + (threadIdx.x >> 5);
  if (er >= ne) return;
  int e = e_base + er;
  int s = es[e], d = ed[e];
  float4 qv = *(const float4*)(q + (size_t)d * 128 + lane * 4);
  float4 kv = *(const float4*)(k + (size_t)s * 128 + lane * 4);
  float4 vv = *(const float4*)(v + (size_t)s * 128 + lane * 4);
  bf16x4 krv = *(const bf16x4*)(kr + (size_t)er * 128 + lane * 4);
  bf16x4 vrv = *(const bf16x4*)(vr + (size_t)er * 128 + lane * 4);
  float acc = qv.x * (kv.x + (float)krv[0]) + qv.y * (kv.y + (float)krv[1]) +
              qv.z * (kv.z + (float)krv[2]) + qv.w * (kv.w + (float)krv[3]);
  bf16x4 vjv;
  vjv[0] = (bf16)(vv.x + (float)vrv[0]);
  vjv[1] = (bf16)(vv.y + (float)vrv[1]);
  vjv[2] = (bf16)(vv.z + (float)vrv[2]);
  vjv[3] = (bf16)(vv.w + (float)vrv[3]);
  *(bf16x4*)(vj + (size_t)e * 128 + lane * 4) = vjv;
  acc += __shfl_xor(acc, 1, 32);
  acc += __shfl_xor(acc, 2, 32);
  if ((lane & 3) == 0) sim[(size_t)e * 8 + (lane >> 2)] = acc * 0.25f;  // HD^-0.5
}

__global__ void seg_max_kernel(const float* __restrict__ sim, const int* __restrict__ ed,
                               unsigned* __restrict__ menc, int E) {
  int idx = blockIdx.x * blockDim.x + threadIdx.x;
  if (idx >= E * 8) return;
  int e = idx >> 3, h = idx & 7;
  atomicMax(&menc[(size_t)ed[e] * 8 + h], fenc(sim[idx]));
}

__global__ void seg_expsum_kernel(const float* __restrict__ sim, const int* __restrict__ ed,
                                  const unsigned* __restrict__ menc, float* __restrict__ z, int E) {
  int idx = blockIdx.x * blockDim.x + threadIdx.x;
  if (idx >= E * 8) return;
  int e = idx >> 3, h = idx & 7;
  int d = ed[e];
  float m = fdec(menc[(size_t)d * 8 + h]);
  atomicAdd(&z[(size_t)d * 8 + h], expf(sim[idx] - m));
}

__global__ void seg_agg_kernel(const float* __restrict__ sim, const int* __restrict__ ed,
                               const unsigned* __restrict__ menc, const float* __restrict__ z,
                               const bf16* __restrict__ vj, float* __restrict__ agg, int E) {
  int lane = threadIdx.x & 31;
  int e    = blockIdx.x * (blockDim.x >> 5) + (threadIdx.x >> 5);
  if (e >= E) return;
  int d = ed[e];
  int h = lane >> 2;
  float m = fdec(menc[(size_t)d * 8 + h]);
  float attn = expf(sim[(size_t)e * 8 + h] - m) / (z[(size_t)d * 8 + h] + 1e-16f);
  bf16x4 vjv = *(const bf16x4*)(vj + (size_t)e * 128 + lane * 4);
#pragma unroll
  for (int i = 0; i < 4; ++i) {
    atomicAdd(&agg[(size_t)d * 128 + lane * 4 + i], attn * (float)vjv[i]);
  }
}

__global__ void gate_combine_kernel(const float* __restrict__ agg, const float* __restrict__ g,
                                    const float* __restrict__ s, bf16* __restrict__ out, int n) {
  int idx = blockIdx.x * blockDim.x + threadIdx.x;
  if (idx >= n) return;
  float a = agg[idx];
  out[idx] = (bf16)(a + g[idx] * (s[idx] - a));
}

// ---------------------------------------------------------------------------
// small utility kernels
// ---------------------------------------------------------------------------
__global__ void f2b_kernel(const float* __restrict__ in, bf16* __restrict__ out, int n) {
  int i = blockIdx.x * blockDim.x + threadIdx.x;
  if (i < n) out[i] = (bf16)in[i];
}
// f32 [slices][K][N] -> bf16 [slices][N][K]  (weights transposed for GEMM B)
__global__ void convT_kernel(const float* __restrict__ in, bf16* __restrict__ out,
                             int slices, int K, int N) {
  int idx = blockIdx.x * blockDim.x + threadIdx.x;
  if (idx >= slices * K * N) return;
  int s = idx / (K * N);
  int r = idx % (K * N);
  int k = r / N, n = r % N;
  out[(size_t)s * N * K + (size_t)n * K + k] = (bf16)in[idx];
}
// W1 f32 [slices][129][128] -> bf16 [slices][128][160] transposed + K-padded
__global__ void padW1T_kernel(const float* __restrict__ in, bf16* __restrict__ out,
                              int slices) {
  int idx = blockIdx.x * blockDim.x + threadIdx.x;
  int tot = slices * 128 * 160;
  if (idx >= tot) return;
  int k = idx % 160;
  int n = (idx / 160) % 128;
  int s = idx / (128 * 160);
  out[idx] = (k < 129) ? (bf16)in[(size_t)s * 129 * 128 + (size_t)k * 128 + n] : (bf16)0.0f;
}
__global__ void fillf_kernel(float* p, float v, int n) {
  int i = blockIdx.x * blockDim.x + threadIdx.x; if (i < n) p[i] = v;
}
__global__ void fillu_kernel(unsigned* p, unsigned v, int n) {
  int i = blockIdx.x * blockDim.x + threadIdx.x; if (i < n) p[i] = v;
}
__global__ void copyf_kernel(const float* __restrict__ in, float* __restrict__ out, int n) {
  int i = blockIdx.x * blockDim.x + threadIdx.x; if (i < n) out[i] = in[i];
}
// in: (R0,R1,128) -> out: (R1,R0,128)
__global__ void transpose_rows_kernel(const float* __restrict__ in, float* __restrict__ out,
                                      int R0, int R1) {
  int idx = blockIdx.x * blockDim.x + threadIdx.x;
  if (idx >= R0 * R1 * 128) return;
  int c = idx & 127;
  int rr = idx >> 7;
  int r1 = rr % R1, r0 = rr / R1;
  out[((size_t)(r1 * R0 + r0)) * 128 + c] = in[(size_t)idx];
}

// ---------------------------------------------------------------------------
// Host-side orchestration
// ---------------------------------------------------------------------------
static inline int ceildiv(int a, int b) { return (a + b - 1) / b; }
static inline int imax(int a, int b) { return a > b ? a : b; }
static inline int imin(int a, int b) { return a < b ? a : b; }
static inline int r64(int a) { return (a + 63) & ~63; }   // row padding for GEMM tiles

static void gemm(hipStream_t st, const bf16* A, const bf16* Bt, const float* bias,
                 float* Cf, bf16* Cb, int M, int N, int K, int ldb, int beta, int act) {
  if (M <= 0) return;
  int useLds = (K <= 256) ? 1 : 0;
  dim3 grid(ceildiv(M, 64), N / 64);
  gemm_kernel<<<grid, dim3(128), 0, st>>>(A, Bt, bias, Cf, Cb, M, N, K, ldb, beta, act, useLds);
}
template <typename Tin>
static void ln128(hipStream_t st, const Tin* in, const float* g, const float* b,
                  const float* res, float* outf, bf16* outb, int rows, int act) {
  if (rows <= 0) return;
  ln128_kernel<Tin><<<ceildiv(rows, 8), dim3(256), 0, st>>>(in, g, b, res, outf, outb, rows, act);
}

struct FourierP {
  const float *freqs, *b1, *g1, *be1, *b2, *g_out, *be_out, *b_out;
  const bf16 *W1, *W2, *Wout;   // transposed [N][K] bf16
  int in_dim;
};
struct AttnP {
  const bf16 *Wq, *Wk, *Wv, *Wkr, *Wvr, *Ws, *Wg, *Wo, *Wf1, *Wf2;  // transposed
  const float *bq, *bv, *bvr, *bs, *bg, *bo, *bf1v, *bf2v;
  const float *lns_g, *lns_b, *lnr_g, *lnr_b, *lnd_g, *lnd_b;
  const float *lnpost_g, *lnpost_b, *lnff_g, *lnff_b, *lnffp_g, *lnffp_b;
  bool bip;
};
struct Scratch {
  bf16 *xs_b, *xd_b, *rn_b, *kr_b, *vr_b, *vj_b, *oib, *hffb, *ff1b, *aggb;
  float *q, *k, *v, *sim, *z, *agg, *gbuf, *sbuf, *obuf, *xnew, *ff2;
  unsigned* menc;
  int ECH;
};

// input index map (recursive insertion-order flatten of setup_inputs())
enum {
  IDX_TYPE_EMB = 0, IDX_XA = 1, IDX_RT = 12, IDX_RP = 23, IDX_RA = 34,
  IDX_TL0 = 45, IDX_TL1 = 73, IDX_PL0 = 101, IDX_PL1 = 131, IDX_AL0 = 161, IDX_AL1 = 189,
  IDX_MAPENC = 217, IDX_MASK = 218, IDX_POSA = 219, IDX_MV = 220, IDX_HEAD = 221,
  IDX_POSPL = 222, IDX_ORIENT = 223, IDX_VEL = 224, IDX_ATYPE = 225,
  IDX_BS = 226, IDX_BPL = 227, IDX_ET = 228, IDX_EP = 229, IDX_EA = 230,
  IDX_COUNT = 231
};

static void run_fourier(hipStream_t st, int ftype, const FourierP& fp, int Ntot,
                        const float* pos_a, const float* head_a, const float* mv,
                        const float* vel, const float* pos_pl, const float* orient,
                        const int* es, const int* ed,
                        const float* type_emb, const int* atype,
                        float* out_f32, bf16* out_bf16,
                        bf16* featb, float* h1, bf16* h1b, float* hsum, bf16* hsumb,
                        int CH) {
  for (int base = 0; base < Ntot; base += CH) {
    int n = imin(CH, Ntot - base);
    for (int i = 0; i < fp.in_dim; ++i) {
      fourier_feat_kernel<<<ceildiv(n, 8), dim3(256), 0, st>>>(
          ftype, i, base, n, fp.freqs + i * NF, pos_a, head_a, mv, vel, pos_pl, orient,
          es, ed, featb);
      gemm(st, featb, fp.W1 + (size_t)i * 128 * 160, fp.b1 + i * 128,
           h1, nullptr, n, 128, 160, 160, 0, 0);
      ln128<float>(st, h1, fp.g1 + i * 128, fp.be1 + i * 128, nullptr, nullptr, h1b, n, 1);
      gemm(st, h1b, fp.W2 + (size_t)i * 128 * 128, fp.b2 + i * 128,
           hsum, nullptr, n, 128, 128, 128, (i > 0) ? 1 : 0, 0);
    }
    if (ftype == 0)
      addcat_kernel<<<ceildiv(n * 128, 256), dim3(256), 0, st>>>(hsum, type_emb, atype, base, n);
    ln128<float>(st, hsum, fp.g_out, fp.be_out, nullptr, nullptr, hsumb, n, 1);
    gemm(st, hsumb, fp.Wout, fp.b_out,
         out_f32 ? out_f32 + (size_t)base * 128 : nullptr,
         out_bf16 ? out_bf16 + (size_t)base * 128 : nullptr,
         n, 128, 128, 128, 0, 0);
  }
}

static void run_attn(hipStream_t st, const AttnP& ap,
                     const float* x_src, int Ns, const float* x_dst, int Nd,
                     const bf16* r_all, int E, const int* es, const int* ed,
                     float* x_out, Scratch& sc) {
  ln128<float>(st, x_src, ap.lns_g, ap.lns_b, nullptr, nullptr, sc.xs_b, Ns, 0);
  const bf16* xdb;
  if (ap.bip) {
    ln128<float>(st, x_dst, ap.lnd_g, ap.lnd_b, nullptr, nullptr, sc.xd_b, Nd, 0);
    xdb = sc.xd_b;
  } else {
    xdb = sc.xs_b;
  }
  gemm(st, xdb,     ap.Wq, ap.bq,   sc.q, nullptr, Nd, 128, 128, 128, 0, 0);
  gemm(st, sc.xs_b, ap.Wk, nullptr, sc.k, nullptr, Ns, 128, 128, 128, 0, 0);
  gemm(st, sc.xs_b, ap.Wv, ap.bv,   sc.v, nullptr, Ns, 128, 128, 128, 0, 0);

  fillu_kernel<<<ceildiv(Nd * 8, 256), 256, 0, st>>>(sc.menc, 0x007FFFFFu, Nd * 8); // enc(-inf)
  fillf_kernel<<<ceildiv(Nd * 8, 256), 256, 0, st>>>(sc.z, 0.0f, Nd * 8);
  fillf_kernel<<<ceildiv(Nd * 128, 256), 256, 0, st>>>(sc.agg, 0.0f, Nd * 128);

  for (int eb = 0; eb < E; eb += sc.ECH) {
    int ne = imin(sc.ECH, E - eb);
    ln128<bf16>(st, r_all + (size_t)eb * 128, ap.lnr_g, ap.lnr_b, nullptr, nullptr, sc.rn_b, ne, 0);
    gemm(st, sc.rn_b, ap.Wkr, nullptr, nullptr, sc.kr_b, ne, 128, 128, 128, 0, 0);
    gemm(st, sc.rn_b, ap.Wvr, ap.bvr, nullptr, sc.vr_b, ne, 128, 128, 128, 0, 0);
    edge_msg_kernel<<<ceildiv(ne, 8), 256, 0, st>>>(sc.q, sc.k, sc.v, sc.kr_b, sc.vr_b,
                                                    es, ed, sc.sim, sc.vj_b, eb, ne);
  }
  if (E > 0) {
    seg_max_kernel<<<ceildiv(E * 8, 256), 256, 0, st>>>(sc.sim, ed, sc.menc, E);
    seg_expsum_kernel<<<ceildiv(E * 8, 256), 256, 0, st>>>(sc.sim, ed, sc.menc, sc.z, E);
    seg_agg_kernel<<<ceildiv(E, 8), 256, 0, st>>>(sc.sim, ed, sc.menc, sc.z, sc.vj_b, sc.agg, E);
  }
  // gate: sigmoid([agg, xd] @ Wg + bg) via split-K on transposed Wg [128][256]
  f2b_kernel<<<ceildiv(Nd * 128, 256), 256, 0, st>>>(sc.agg, sc.aggb, Nd * 128);
  gemm(st, sc.aggb, ap.Wg,       nullptr, sc.gbuf, nullptr, Nd, 128, 128, 256, 0, 0);
  gemm(st, xdb,     ap.Wg + 128, ap.bg,   sc.gbuf, nullptr, Nd, 128, 128, 256, 1, 2);
  gemm(st, xdb,     ap.Ws, ap.bs, sc.sbuf, nullptr, Nd, 128, 128, 128, 0, 0);
  gate_combine_kernel<<<ceildiv(Nd * 128, 256), 256, 0, st>>>(sc.agg, sc.gbuf, sc.sbuf,
                                                              sc.oib, Nd * 128);
  gemm(st, sc.oib, ap.Wo, ap.bo, sc.obuf, nullptr, Nd, 128, 128, 128, 0, 0);
  // x = x_dst + LN(out)
  ln128<float>(st, sc.obuf, ap.lnpost_g, ap.lnpost_b, x_dst, sc.xnew, nullptr, Nd, 0);
  // FFN
  ln128<float>(st, sc.xnew, ap.lnff_g, ap.lnff_b, nullptr, nullptr, sc.hffb, Nd, 0);
  gemm(st, sc.hffb, ap.Wf1, ap.bf1v, nullptr, sc.ff1b, Nd, 512, 128, 128, 0, 1);
  gemm(st, sc.ff1b, ap.Wf2, ap.bf2v, sc.ff2, nullptr, Nd, 128, 512, 512, 0, 0);
  ln128<float>(st, sc.ff2, ap.lnffp_g, ap.lnffp_b, sc.xnew, x_out, nullptr, Nd, 0);
}

extern "C" void kernel_launch(void* const* d_in, const int* in_sizes, int n_in,
                              void* d_out, int out_size, void* d_ws, size_t ws_size,
                              hipStream_t stream) {
  if (n_in < IDX_COUNT) return;

  const int NT  = AA * TT;   // 6400 agent-time nodes (multiple of 64)
  const int NPL = PP * TT;   // 12800 map nodes (multiple of 64)
  const int Et = in_sizes[IDX_ET] / 2;
  const int Ep = in_sizes[IDX_EP] / 2;
  const int Ea = in_sizes[IDX_EA] / 2;
  const int Emax = imax(imax(Et, Ep), Ea);
  const int CH = 16384, ECH = 65536;

  char* base = (char*)d_ws;
  size_t off = 0;
  auto galloc = [&](size_t bytes) -> void* {
    void* p = base + off;
    off = (off + bytes + 255) & ~(size_t)255;
    return p;
  };
  // convert+transpose weight f32 [K][N] -> bf16 [N][K]
  auto convT = [&](int idx, int slices, int K, int N) -> bf16* {
    int n = slices * K * N;
    bf16* p = (bf16*)galloc((size_t)n * sizeof(bf16));
    convT_kernel<<<ceildiv(n, 256), 256, 0, stream>>>((const float*)d_in[idx], p, slices, K, N);
    return p;
  };

  auto mkFourier = [&](int b, int in_dim) -> FourierP {
    FourierP fp;
    fp.in_dim = in_dim;
    fp.freqs  = (const float*)d_in[b + 0];
    fp.b1     = (const float*)d_in[b + 2];
    fp.g1     = (const float*)d_in[b + 3];
    fp.be1    = (const float*)d_in[b + 4];
    fp.b2     = (const float*)d_in[b + 6];
    fp.g_out  = (const float*)d_in[b + 7];
    fp.be_out = (const float*)d_in[b + 8];
    fp.b_out  = (const float*)d_in[b + 10];
    bf16* w1  = (bf16*)galloc((size_t)in_dim * 128 * 160 * sizeof(bf16));
    padW1T_kernel<<<ceildiv(in_dim * 128 * 160, 256), 256, 0, stream>>>(
        (const float*)d_in[b + 1], w1, in_dim);
    fp.W1   = w1;
    fp.W2   = convT(b + 5, in_dim, 128, 128);
    fp.Wout = convT(b + 9, 1, 128, 128);
    return fp;
  };
  auto mkAttn = [&](int b, bool bip) -> AttnP {
    AttnP ap{};
    ap.Wq  = convT(b + 0, 1, 128, 128);  ap.bq   = (const float*)d_in[b + 1];
    ap.Wk  = convT(b + 2, 1, 128, 128);
    ap.Wv  = convT(b + 3, 1, 128, 128);  ap.bv   = (const float*)d_in[b + 4];
    ap.Wkr = convT(b + 5, 1, 128, 128);
    ap.Wvr = convT(b + 6, 1, 128, 128);  ap.bvr  = (const float*)d_in[b + 7];
    ap.Ws  = convT(b + 8, 1, 128, 128);  ap.bs   = (const float*)d_in[b + 9];
    ap.Wg  = convT(b + 10, 1, 256, 128); ap.bg   = (const float*)d_in[b + 11];
    ap.Wo  = convT(b + 12, 1, 128, 128); ap.bo   = (const float*)d_in[b + 13];
    ap.Wf1 = convT(b + 14, 1, 128, 512); ap.bf1v = (const float*)d_in[b + 15];
    ap.Wf2 = convT(b + 16, 1, 512, 128); ap.bf2v = (const float*)d_in[b + 17];
    ap.lns_g = (const float*)d_in[b + 18];    ap.lns_b = (const float*)d_in[b + 19];
    ap.lnr_g = (const float*)d_in[b + 20];    ap.lnr_b = (const float*)d_in[b + 21];
    ap.lnpost_g = (const float*)d_in[b + 22]; ap.lnpost_b = (const float*)d_in[b + 23];
    ap.lnff_g = (const float*)d_in[b + 24];   ap.lnff_b = (const float*)d_in[b + 25];
    ap.lnffp_g = (const float*)d_in[b + 26];  ap.lnffp_b = (const float*)d_in[b + 27];
    ap.bip = bip;
    if (bip) { ap.lnd_g = (const float*)d_in[b + 28]; ap.lnd_b = (const float*)d_in[b + 29]; }
    return ap;
  };

  FourierP fx  = mkFourier(IDX_XA, 4);
  FourierP frt = mkFourier(IDX_RT, 4);
  FourierP frp = mkFourier(IDX_RP, 3);
  FourierP fra = mkFourier(IDX_RA, 3);
  AttnP tl[2] = { mkAttn(IDX_TL0, false), mkAttn(IDX_TL1, false) };
  AttnP pl[2] = { mkAttn(IDX_PL0, true),  mkAttn(IDX_PL1, true)  };
  AttnP al[2] = { mkAttn(IDX_AL0, false), mkAttn(IDX_AL1, false) };

  // --- activations / intermediates (rows padded to x64 for guard-free GEMM) --
  float* x0   = (float*)galloc((size_t)NT * 128 * 4);
  float* x1   = (float*)galloc((size_t)NT * 128 * 4);
  float* x2   = (float*)galloc((size_t)NT * 128 * 4);
  float* x_pl = (float*)galloc((size_t)NPL * 128 * 4);
  bf16* rt_b = (bf16*)galloc((size_t)r64(imax(Et, 1)) * 128 * 2);
  bf16* rp_b = (bf16*)galloc((size_t)r64(imax(Ep, 1)) * 128 * 2);
  bf16* ra_b = (bf16*)galloc((size_t)r64(imax(Ea, 1)) * 128 * 2);
  // fourier scratch
  bf16*  featb = (bf16*)galloc((size_t)CH * 160 * 2);
  float* h1    = (float*)galloc((size_t)CH * 128 * 4);
  bf16*  h1b   = (bf16*)galloc((size_t)CH * 128 * 2);
  float* hsum  = (float*)galloc((size_t)CH * 128 * 4);
  bf16*  hsumb = (bf16*)galloc((size_t)CH * 128 * 2);
  // attention scratch
  Scratch sc;
  sc.ECH  = ECH;
  sc.xs_b = (bf16*)galloc((size_t)NPL * 128 * 2);
  sc.xd_b = (bf16*)galloc((size_t)NT * 128 * 2);
  sc.rn_b = (bf16*)galloc((size_t)ECH * 128 * 2);
  sc.kr_b = (bf16*)galloc((size_t)ECH * 128 * 2);
  sc.vr_b = (bf16*)galloc((size_t)ECH * 128 * 2);
  sc.vj_b = (bf16*)galloc((size_t)r64(imax(Emax, 1)) * 128 * 2);
  sc.sim  = (float*)galloc((size_t)r64(imax(Emax, 1)) * 8 * 4);
  sc.menc = (unsigned*)galloc((size_t)NT * 8 * 4);
  sc.z    = (float*)galloc((size_t)NT * 8 * 4);
  sc.agg  = (float*)galloc((size_t)NT * 128 * 4);
  sc.aggb = (bf16*)galloc((size_t)NT * 128 * 2);
  sc.gbuf = (float*)galloc((size_t)NT * 128 * 4);
  sc.sbuf = (float*)galloc((size_t)NT * 128 * 4);
  sc.oib  = (bf16*)galloc((size_t)NT * 128 * 2);
  sc.obuf = (float*)galloc((size_t)NT * 128 * 4);
  sc.xnew = (float*)galloc((size_t)NT * 128 * 4);
  sc.hffb = (bf16*)galloc((size_t)NT * 128 * 2);
  sc.ff1b = (bf16*)galloc((size_t)NT * 512 * 2);
  sc.ff2  = (float*)galloc((size_t)NT * 128 * 4);
  (void)ws_size;

  const float* pos_a  = (const float*)d_in[IDX_POSA];
  const float* head_a = (const float*)d_in[IDX_HEAD];
  const float* mv     = (const float*)d_in[IDX_MV];
  const float* vel    = (const float*)d_in[IDX_VEL];
  const float* pos_pl = (const float*)d_in[IDX_POSPL];
  const float* orient = (const float*)d_in[IDX_ORIENT];
  const int*   atype  = (const int*)d_in[IDX_ATYPE];
  const int* et = (const int*)d_in[IDX_ET];
  const int* ep = (const int*)d_in[IDX_EP];
  const int* ea = (const int*)d_in[IDX_EA];
  const float* type_emb = (const float*)d_in[IDX_TYPE_EMB];

  // x_pl = map_enc_x_pl (P,T,H) -> (T,P,H)
  transpose_rows_kernel<<<ceildiv(NPL * 128, 256), 256, 0, stream>>>(
      (const float*)d_in[IDX_MAPENC], x_pl, PP, TT);

  // Fourier embeddings
  run_fourier(stream, 0, fx, NT, pos_a, head_a, mv, vel, pos_pl, orient,
              nullptr, nullptr, type_emb, atype, x0, nullptr,
              featb, h1, h1b, hsum, hsumb, CH);
  run_fourier(stream, 1, frt, Et, pos_a, head_a, mv, vel, pos_pl, orient,
              et, et + Et, nullptr, nullptr, nullptr, rt_b,
              featb, h1, h1b, hsum, hsumb, CH);
  run_fourier(stream, 2, frp, Ep, pos_a, head_a, mv, vel, pos_pl, orient,
              ep, ep + Ep, nullptr, nullptr, nullptr, rp_b,
              featb, h1, h1b, hsum, hsumb, CH);
  run_fourier(stream, 3, fra, Ea, pos_a, head_a, mv, vel, pos_pl, orient,
              ea, ea + Ea, nullptr, nullptr, nullptr, ra_b,
              featb, h1, h1b, hsum, hsumb, CH);

  // attention stack
  float* xcur = x0;  // (A,T) layout
  for (int l = 0; l < 2; ++l) {
    run_attn(stream, tl[l], xcur, NT, xcur, NT, rt_b, Et, et, et + Et, x1, sc);
    transpose_rows_kernel<<<ceildiv(NT * 128, 256), 256, 0, stream>>>(x1, x2, AA, TT); // ->(T,A)
    run_attn(stream, pl[l], x_pl, NPL, x2, NT, rp_b, Ep, ep, ep + Ep, x1, sc);
    run_attn(stream, al[l], x1, NT, x1, NT, ra_b, Ea, ea, ea + Ea, x2, sc);
    transpose_rows_kernel<<<ceildiv(NT * 128, 256), 256, 0, stream>>>(x2, x0, TT, AA); // ->(A,T)
    xcur = x0;
  }

  int n_out = imin(out_size, NT * 128);
  copyf_kernel<<<ceildiv(n_out, 256), 256, 0, stream>>>(xcur, (float*)d_out, n_out);
}
